// CausalMultiheadSelfAttention_36438502539807
// MI455X (gfx1250) — matrix operand order repaired and checked by reading, not compiled, not run on hardware
//
#include <hip/hip_runtime.h>

#define DM    1024
#define SEQ   2048
#define BATCH 4
#define NH    16
#define HD    64
#define MTOT  (BATCH * SEQ)   // 8192

typedef __attribute__((ext_vector_type(16))) __bf16 v16bf;
typedef __attribute__((ext_vector_type(8)))  __bf16 v8bf;
typedef __attribute__((ext_vector_type(8)))  float  v8f;

// ---------------------------------------------------------------------------
// Fragment loaders following CDNA5 WMMA VGPR layouts (CDNA5 ISA)
// A 16x32 (MxK) bf16: lane m=L%16, half kh=L/16; elem[2v,2v+1] = K = g*16+kh*8+vv*2
// B 32x16 (KxN) bf16: lane n=L%16, half kh=L/16; elem[i]       = K = (i/8)*16+kh*8+i%8
// C/D 16x16 f32:      lane n=L%16; VGPR v -> M = (L/16)*8 + v
// ---------------------------------------------------------------------------

// A fragment from a bf16 row-major matrix
__device__ inline v16bf load_a_bf16(const __bf16* __restrict__ base, int lda, int k0, int lane) {
  const __bf16* row = base + (size_t)(lane & 15) * lda + k0;
  const int kh = (lane >> 4) * 8;
  v8bf lo = *(const v8bf*)(row + kh);        // K = kh*8 + 0..7      (group 0)
  v8bf hi = *(const v8bf*)(row + 16 + kh);   // K = 16 + kh*8 + 0..7 (group 1)
  v16bf a;
#pragma unroll
  for (int i = 0; i < 8; ++i) { a[i] = lo[i]; a[8 + i] = hi[i]; }
  return a;
}

// B fragment where B(k,n) = M[n][coff + k] for bf16 row-major M (two 8-K groups, same order as A)
__device__ inline v16bf load_b_bf16(const __bf16* __restrict__ base, int row_stride, int coff, int lane) {
  const __bf16* p = base + (size_t)(lane & 15) * row_stride + coff + (lane >> 4) * 8;
  v8bf lo = *(const v8bf*)(p);
  v8bf hi = *(const v8bf*)(p + 16);
  v16bf b;
#pragma unroll
  for (int i = 0; i < 8; ++i) { b[i] = lo[i]; b[8 + i] = hi[i]; }
  return b;
}

// ---------------------------------------------------------------------------
// Kernel 0: one-time f32 -> bf16 conversion (8 elems / thread, b128 in, b128/2 out)
// ---------------------------------------------------------------------------
__global__ __launch_bounds__(256) void cvt_bf16_kernel(const float* __restrict__ src,
                                                       __bf16* __restrict__ dst, int n8) {
  const int i = blockIdx.x * 256 + threadIdx.x;
  if (i >= n8) return;
  const float4* s = (const float4*)src;
  float4 f0 = s[2 * i], f1 = s[2 * i + 1];
  v8bf o;
  o[0] = (__bf16)f0.x; o[1] = (__bf16)f0.y; o[2] = (__bf16)f0.z; o[3] = (__bf16)f0.w;
  o[4] = (__bf16)f1.x; o[5] = (__bf16)f1.y; o[6] = (__bf16)f1.z; o[7] = (__bf16)f1.w;
  *((v8bf*)dst + i) = o;
}

// ---------------------------------------------------------------------------
// Kernel 1: fused QKV projection (pure bf16 WMMA, f32 accum) + RoPE + layouts
//   Q -> [B,H,S,64] bf16 (scaled by 1/8), K -> [B,H,S,64] bf16, V -> [B,H,64,S]
// grid = (MTOT/32, 6), block = 256 (8 waves); wave owns one 32x64 strip
// ---------------------------------------------------------------------------
__global__ __launch_bounds__(256) void qkv_rope_kernel(
    const __bf16* __restrict__ xb, const int* __restrict__ tpos,
    const __bf16* __restrict__ wqb, const __bf16* __restrict__ wkb,
    const __bf16* __restrict__ wvb,
    __bf16* __restrict__ Qw, __bf16* __restrict__ Kw, __bf16* __restrict__ Vtw)
{
  const int lane  = threadIdx.x & 31;
  const int wave  = threadIdx.x >> 5;
  const int mtile = blockIdx.x * 32;
  const int strip = blockIdx.y * 8 + wave;   // 0..47
  const int mat   = strip >> 4;              // 0=Q, 1=K, 2=V
  const int n0    = (strip & 15) * 64;       // output-feature base (one head)
  const __bf16* W = (mat == 0) ? wqb : ((mat == 1) ? wkb : wvb);

  const __bf16* a0base = xb + (size_t)mtile * DM;
  const __bf16* a1base = a0base + (size_t)16 * DM;

  v8f acc[2][4];
#pragma unroll
  for (int mt = 0; mt < 2; ++mt)
#pragma unroll
    for (int t = 0; t < 4; ++t)
#pragma unroll
      for (int i = 0; i < 8; ++i) acc[mt][t][i] = 0.0f;

  for (int k0 = 0; k0 < DM; k0 += 32) {
    __builtin_prefetch(a0base + (size_t)(lane & 15) * DM + k0 + 64, 0, 1);
    v16bf a0 = load_a_bf16(a0base, DM, k0, lane);
    v16bf a1 = load_a_bf16(a1base, DM, k0, lane);
#pragma unroll
    for (int t = 0; t < 4; ++t) {
      v16bf b = load_b_bf16(W + (size_t)(n0 + t * 16) * DM, DM, k0, lane);
      acc[0][t] = __builtin_amdgcn_wmma_f32_16x16x32_bf16(false, a0, false, b,
                                                          (short)0, acc[0][t], false, false);
      acc[1][t] = __builtin_amdgcn_wmma_f32_16x16x32_bf16(false, a1, false, b,
                                                          (short)0, acc[1][t], false, false);
    }
  }

  const int hf = lane >> 4;         // which 8-row half this lane's D regs hold
  const int nl = lane & 15;
  const int bidx = mtile / SEQ;     // batch (tiles never straddle batches)
  const int h    = n0 / HD;         // head for this strip

#pragma unroll
  for (int mt = 0; mt < 2; ++mt) {
    const int s0 = ((mtile + mt * 16) % SEQ) + hf * 8;
    if (mat == 2) {
      // V: store transposed [B,H,64,S]; per lane 8 consecutive s -> 16B store
#pragma unroll
      for (int t = 0; t < 4; ++t) {
        const int d = t * 16 + nl;
        v8bf pk;
#pragma unroll
        for (int v = 0; v < 8; ++v) pk[v] = (__bf16)acc[mt][t][v];
        *(v8bf*)(Vtw + (((size_t)(bidx * NH + h) * HD + d) * SEQ + s0)) = pk;
      }
    } else {
      __bf16* dst = (mat == 0) ? Qw : Kw;
      const float scale = (mat == 0) ? 0.125f : 1.0f; // fold 1/sqrt(64) into Q
#pragma unroll
      for (int t = 0; t < 4; ++t) {
        const int d = t * 16 + nl;
        const int p = d >> 1;
        const bool even = (d & 1) == 0;
        // thetas_p = 10000^(-p/32); ln(10000)/32 = 0.2878231366...
        const float inv = __expf(-(float)p * 0.2878231366242557f);
#pragma unroll
        for (int v = 0; v < 8; ++v) {
          const int s = s0 + v;
          const float pos = (float)tpos[s];
          const float ang = pos * inv;
          const float c = __cosf(ang), sn = __sinf(ang);
          const float val = acc[mt][t][v];
          const float partner = __shfl_xor(val, 1, 32); // paired even/odd feature
          float r = even ? (val * c - partner * sn) : (partner * sn + val * c);
          r *= scale;
          dst[(((size_t)(bidx * NH + h) * SEQ + s) * HD) + d] = (__bf16)r;
        }
      }
    }
  }
}

// ---------------------------------------------------------------------------
// Kernel 2: causal flash attention. One wave per (b,h,32-query tile).
// Per 32-key step: 8 WMMA (S = Q K^T) + mask + online softmax + LDS transpose
// of P (D-layout -> A-layout) + 8 WMMA (O += P V); K/V frags shared by 2 tiles.
// grid = (SEQ/32, B*H), block = 32
// ---------------------------------------------------------------------------
__global__ __launch_bounds__(32) void flash_attn_kernel(
    const __bf16* __restrict__ Qw, const __bf16* __restrict__ Kw,
    const __bf16* __restrict__ Vtw, __bf16* __restrict__ Aout)
{
  __shared__ __bf16 Pl[32 * 32];                 // P staged for layout swap

  const int lane  = threadIdx.x & 31;
  const int qbase = blockIdx.x * 32;
  const int bh    = blockIdx.y;                  // b*NH + h
  const int bidx  = bh / NH, h = bh % NH;
  const int hf = lane >> 4, nl = lane & 15;

  // Q A-fragments for both query tiles (K-dim halves 0..31 / 32..63)
  v16bf aq[2][2];
#pragma unroll
  for (int mt = 0; mt < 2; ++mt) {
    const __bf16* qptr = Qw + ((size_t)bh * SEQ + qbase + mt * 16) * HD;
    aq[mt][0] = load_a_bf16(qptr, HD, 0,  lane);
    aq[mt][1] = load_a_bf16(qptr, HD, 32, lane);
  }

  v8f o[2][4];
  float rowM[2][8], rowS[2][8];
#pragma unroll
  for (int mt = 0; mt < 2; ++mt) {
#pragma unroll
    for (int t = 0; t < 4; ++t)
#pragma unroll
      for (int i = 0; i < 8; ++i) o[mt][t][i] = 0.0f;
#pragma unroll
    for (int v = 0; v < 8; ++v) { rowM[mt][v] = -1e30f; rowS[mt][v] = 0.0f; }
  }

  v8f zeroC;
#pragma unroll
  for (int i = 0; i < 8; ++i) zeroC[i] = 0.0f;

  const int steps = blockIdx.x + 1;              // keys 0..qbase+31, 32 at a time
  for (int it = 0; it < steps; ++it) {
    const int kb = it * 32;

    // ---- K fragments for keys [kb, kb+32), shared by both query tiles -----
    const __bf16* kp0 = Kw + ((size_t)bh * SEQ + kb) * HD;
    const __bf16* kp1 = kp0 + (size_t)16 * HD;
    const v16bf bk00 = load_b_bf16(kp0, HD, 0,  lane);
    const v16bf bk01 = load_b_bf16(kp0, HD, 32, lane);
    const v16bf bk10 = load_b_bf16(kp1, HD, 0,  lane);
    const v16bf bk11 = load_b_bf16(kp1, HD, 32, lane);

#pragma unroll
    for (int mt = 0; mt < 2; ++mt) {
      v8f sc0 = __builtin_amdgcn_wmma_f32_16x16x32_bf16(false, aq[mt][0], false, bk00, (short)0, zeroC, false, false);
      sc0     = __builtin_amdgcn_wmma_f32_16x16x32_bf16(false, aq[mt][1], false, bk01, (short)0, sc0,   false, false);
      v8f sc1 = __builtin_amdgcn_wmma_f32_16x16x32_bf16(false, aq[mt][0], false, bk10, (short)0, zeroC, false, false);
      sc1     = __builtin_amdgcn_wmma_f32_16x16x32_bf16(false, aq[mt][1], false, bk11, (short)0, sc1,   false, false);

      // ---- causal mask + online softmax (per row) -------------------------
#pragma unroll
      for (int v = 0; v < 8; ++v) {
        const int qrow = qbase + mt * 16 + hf * 8 + v;
        float s0 = (kb + nl      <= qrow) ? sc0[v] : -1e30f;
        float s1 = (kb + 16 + nl <= qrow) ? sc1[v] : -1e30f;
        float mx = fmaxf(s0, s1);
#pragma unroll
        for (int off = 8; off >= 1; off >>= 1) mx = fmaxf(mx, __shfl_xor(mx, off, 32));
        const float nm = fmaxf(rowM[mt][v], mx);
        const float al = __expf(rowM[mt][v] - nm);
        const float p0 = __expf(s0 - nm);
        const float p1 = __expf(s1 - nm);
        float ps = p0 + p1;
#pragma unroll
        for (int off = 8; off >= 1; off >>= 1) ps += __shfl_xor(ps, off, 32);
        rowS[mt][v] = rowS[mt][v] * al + ps;
        rowM[mt][v] = nm;
#pragma unroll
        for (int t = 0; t < 4; ++t) o[mt][t][v] *= al;
        Pl[(mt * 16 + hf * 8 + v) * 32 + nl]      = (__bf16)p0;
        Pl[(mt * 16 + hf * 8 + v) * 32 + 16 + nl] = (__bf16)p1;
      }
    }
    __syncthreads();

    // ---- O += P V : V fragments shared by both query tiles ----------------
    v16bf bv[4];
#pragma unroll
    for (int t = 0; t < 4; ++t)
      bv[t] = load_b_bf16(Vtw + ((size_t)bh * HD + t * 16) * SEQ + kb, SEQ, 0, lane);
#pragma unroll
    for (int mt = 0; mt < 2; ++mt) {
      const v16bf ap = load_a_bf16(&Pl[mt * 16 * 32], 32, 0, lane);
#pragma unroll
      for (int t = 0; t < 4; ++t)
        o[mt][t] = __builtin_amdgcn_wmma_f32_16x16x32_bf16(false, ap, false, bv[t],
                                                           (short)0, o[mt][t], false, false);
    }
    __syncthreads();
  }

  // ---- normalize + store to [B,S,1024] bf16 (col = h*64 + d) --------------
#pragma unroll
  for (int mt = 0; mt < 2; ++mt)
#pragma unroll
    for (int t = 0; t < 4; ++t)
#pragma unroll
      for (int v = 0; v < 8; ++v) {
        const int srow = qbase + mt * 16 + hf * 8 + v;
        const float r = o[mt][t][v] / rowS[mt][v];
        Aout[((size_t)bidx * SEQ + srow) * DM + h * HD + t * 16 + nl] = (__bf16)r;
      }
}

// ---------------------------------------------------------------------------
// Kernel 3: output projection  out = attn(bf16) x Wo^T(bf16) -> f32
// grid = (MTOT/32, 2), block = 256 (8 waves); wave owns a 32x64 strip
// ---------------------------------------------------------------------------
__global__ __launch_bounds__(256) void out_proj_kernel(
    const __bf16* __restrict__ A, const __bf16* __restrict__ wob, float* __restrict__ out)
{
  const int lane  = threadIdx.x & 31;
  const int wave  = threadIdx.x >> 5;
  const int mtile = blockIdx.x * 32;
  const int strip = blockIdx.y * 8 + wave;   // 0..15
  const int n0    = strip * 64;

  const __bf16* a0base = A + (size_t)mtile * DM;
  const __bf16* a1base = a0base + (size_t)16 * DM;

  v8f acc[2][4];
#pragma unroll
  for (int mt = 0; mt < 2; ++mt)
#pragma unroll
    for (int t = 0; t < 4; ++t)
#pragma unroll
      for (int i = 0; i < 8; ++i) acc[mt][t][i] = 0.0f;

  for (int k0 = 0; k0 < DM; k0 += 32) {
    __builtin_prefetch(a0base + (size_t)(lane & 15) * DM + k0 + 64, 0, 1);
    v16bf a0 = load_a_bf16(a0base, DM, k0, lane);
    v16bf a1 = load_a_bf16(a1base, DM, k0, lane);
#pragma unroll
    for (int t = 0; t < 4; ++t) {
      v16bf b = load_b_bf16(wob + (size_t)(n0 + t * 16) * DM, DM, k0, lane);
      acc[0][t] = __builtin_amdgcn_wmma_f32_16x16x32_bf16(false, a0, false, b,
                                                          (short)0, acc[0][t], false, false);
      acc[1][t] = __builtin_amdgcn_wmma_f32_16x16x32_bf16(false, a1, false, b,
                                                          (short)0, acc[1][t], false, false);
    }
  }

  const int hf = lane >> 4, nl = lane & 15;
#pragma unroll
  for (int mt = 0; mt < 2; ++mt)
#pragma unroll
    for (int t = 0; t < 4; ++t)
#pragma unroll
      for (int v = 0; v < 8; ++v)
        out[(size_t)(mtile + mt * 16 + hf * 8 + v) * DM + n0 + t * 16 + nl] = acc[mt][t][v];
}

// ---------------------------------------------------------------------------
extern "C" void kernel_launch(void* const* d_in, const int* in_sizes, int n_in,
                              void* d_out, int out_size, void* d_ws, size_t ws_size,
                              hipStream_t stream) {
  const float* x    = (const float*)d_in[0];
  const int*   tpos = (const int*)  d_in[1];
  const float* wq   = (const float*)d_in[2];
  const float* wk   = (const float*)d_in[3];
  const float* wv   = (const float*)d_in[4];
  const float* wo   = (const float*)d_in[5];
  float* out = (float*)d_out;

  const size_t tenB = (size_t)BATCH * NH * SEQ * HD * 2;  // 16 MB per bf16 tensor
  const size_t wB   = (size_t)DM * DM * 2;                // 2 MB per bf16 weight
  char* ws = (char*)d_ws;
  __bf16* Qw   = (__bf16*)(ws);
  __bf16* Kw   = (__bf16*)(ws + tenB);
  __bf16* Vtw  = (__bf16*)(ws + 2 * tenB);
  __bf16* Aout = (__bf16*)(ws + 3 * tenB);
  __bf16* xb   = (__bf16*)(ws + 4 * tenB);
  __bf16* wqb  = (__bf16*)(ws + 5 * tenB);
  __bf16* wkb  = (__bf16*)(ws + 5 * tenB + wB);
  __bf16* wvb  = (__bf16*)(ws + 5 * tenB + 2 * wB);
  __bf16* wob  = (__bf16*)(ws + 5 * tenB + 3 * wB);

  // one-time f32 -> bf16 conversions
  const int nx8 = MTOT * DM / 8, nw8 = DM * DM / 8;
  cvt_bf16_kernel<<<nx8 / 256, 256, 0, stream>>>(x,  xb,  nx8);
  cvt_bf16_kernel<<<nw8 / 256, 256, 0, stream>>>(wq, wqb, nw8);
  cvt_bf16_kernel<<<nw8 / 256, 256, 0, stream>>>(wk, wkb, nw8);
  cvt_bf16_kernel<<<nw8 / 256, 256, 0, stream>>>(wv, wvb, nw8);
  cvt_bf16_kernel<<<nw8 / 256, 256, 0, stream>>>(wo, wob, nw8);

  dim3 b1(256), g1(MTOT / 32, 6);         // 48 strips (Q,K,V x 16) / 8 waves
  qkv_rope_kernel<<<g1, b1, 0, stream>>>(xb, tpos, wqb, wkb, wvb, Qw, Kw, Vtw);

  dim3 b2(32), g2(SEQ / 32, BATCH * NH);
  flash_attn_kernel<<<g2, b2, 0, stream>>>(Qw, Kw, Vtw, Aout);

  dim3 b3(256), g3(MTOT / 32, 2);         // 16 strips / 8 waves
  out_proj_kernel<<<g3, b3, 0, stream>>>(Aout, wob, out);
}
